// FieldDiffusion_3736621547599
// MI455X (gfx1250) — matrix operand order repaired and checked by reading, not compiled, not run on hardware
//
#include <hip/hip_runtime.h>
#include <hip/hip_bf16.h>
#include <math.h>
#include <stdint.h>

typedef __attribute__((ext_vector_type(16))) __bf16 v16bf;
typedef __attribute__((ext_vector_type(8)))  __bf16 v8bf;
typedef __attribute__((ext_vector_type(8)))  float  v8f;
typedef __attribute__((ext_vector_type(4)))  float  v4f;

#define DIM   2048
#define SEQ   4096
#define MTOT  16384     // BATCH * SEQ
#define MT    256       // workgroup M tile
#define NT    128       // workgroup N tile
#define KT    32        // K step (one bf16 WMMA deep)
#define LDSTR 40        // padded LDS row stride in bf16 elems (80B, 16B-aligned)
#define NITER (DIM / KT)

// ---- CDNA5 async global->LDS copy (ASYNCcnt path), per bridge doc use asm ----
__device__ __forceinline__ uint32_t lds_off(const void* p) {
    // generic LDS pointer: addr[31:0] is the LDS byte offset (aperture in MSBs)
    return (uint32_t)(uintptr_t)p;
}
__device__ __forceinline__ void async_copy_b128(const void* gsrc, uint32_t ldst) {
    asm volatile("global_load_async_to_lds_b128 %0, %1, off"
                 :: "v"(ldst), "v"((uint64_t)(uintptr_t)gsrc));
}
__device__ __forceinline__ void wait_async0() {
    asm volatile("s_wait_asynccnt 0" ::: "memory");
}

// ---------------------------------------------------------------------------
// Prep: h fp32 -> bf16
// ---------------------------------------------------------------------------
__global__ void __launch_bounds__(256)
prep_h_kernel(const float* __restrict__ h, __bf16* __restrict__ Hbf) {
    size_t i = ((size_t)blockIdx.x * 256 + threadIdx.x) * 8;
    v4f f0 = *(const v4f*)(h + i);
    v4f f1 = *(const v4f*)(h + i + 4);
    v8bf p;
    #pragma unroll
    for (int j = 0; j < 4; ++j) { p[j] = (__bf16)f0[j]; p[4 + j] = (__bf16)f1[j]; }
    *(v8bf*)(Hbf + i) = p;
}

// ---------------------------------------------------------------------------
// Prep: W1 fp32 -> bf16 row-major copy (W1bf) + transposed copy (W1T)
// ---------------------------------------------------------------------------
__global__ void __launch_bounds__(256)
prep_w1_kernel(const float* __restrict__ W1,
               __bf16* __restrict__ W1bf,
               __bf16* __restrict__ W1T) {
    int i = blockIdx.x * 256 + threadIdx.x;
    int e = i >> 11;
    int d = i & (DIM - 1);
    float w = W1[i];
    W1bf[i] = (__bf16)w;
    W1T[(size_t)d * DIM + e] = (__bf16)w;
}

// ---------------------------------------------------------------------------
// Prep: c2[e] = sum_r W2[r][e]
// ---------------------------------------------------------------------------
__global__ void __launch_bounds__(256)
prep_c2_kernel(const float* __restrict__ W2, float* __restrict__ c2) {
    int e = blockIdx.x * 256 + threadIdx.x;
    float s = 0.0f;
    for (int r = 0; r < DIM; ++r) s += W2[(size_t)r * DIM + e];
    c2[e] = s;
}

// ---------------------------------------------------------------------------
// GEMM: C = A @ BT^T, A: MxK bf16 row-major, BT: NxK bf16 row-major.
// Workgroup 256x128, 8 waves as 4x2, each wave 64x64 = 4x4 WMMA tiles.
// LDS double-buffered, filled with global_load_async_to_lds_b128,
// one barrier per K step.
// MODE 0: epilogue -> G = sech^2(z+b1)*c2   (bf16)
// MODE 1: epilogue -> out = phi + DT*(D*lap - C)  (fp32)
// ---------------------------------------------------------------------------
template <int MODE>
__global__ void __launch_bounds__(256)
gemm_kernel(const __bf16* __restrict__ A,
            const __bf16* __restrict__ BT,
            const float*  __restrict__ b1,
            const float*  __restrict__ c2,
            __bf16*       __restrict__ G,
            const float*  __restrict__ h,
            float*        __restrict__ out)
{
    __shared__ __align__(16) __bf16 lA [2][MT * LDSTR];   // 40 KiB
    __shared__ __align__(16) __bf16 lBT[2][NT * LDSTR];   // 20 KiB

    const int tid  = threadIdx.x;
    const int lane = tid & 31;
    const int wave = tid >> 5;
    const int wm = (wave >> 1) * 64;
    const int wn = (wave & 1)  * 64;
    const int n0 = blockIdx.x * NT;
    const int m0 = blockIdx.y * MT;

    const v8f vzero = {0.f, 0.f, 0.f, 0.f, 0.f, 0.f, 0.f, 0.f};
    v8f acc[4][4];
    #pragma unroll
    for (int i = 0; i < 4; ++i)
        #pragma unroll
        for (int j = 0; j < 4; ++j) acc[i][j] = vzero;

    // staging: A -> thread t copies row (m0+t), 64B; BT -> 2 threads per row, 32B each
    const int bn  = tid >> 1;
    const int bkh = (tid & 1) * 16;
    const __bf16* aRow  = A  + (size_t)(m0 + tid) * DIM;
    const __bf16* btRow = BT + (size_t)(n0 + bn) * DIM + bkh;

    uint32_t oA[2], oB[2];
    oA[0] = lds_off(&lA[0][tid * LDSTR]);
    oA[1] = lds_off(&lA[1][tid * LDSTR]);
    oB[0] = lds_off(&lBT[0][bn * LDSTR + bkh]);
    oB[1] = lds_off(&lBT[1][bn * LDSTR + bkh]);

    // fragment coords (documented 16-bit WMMA layouts, wave32)
    const int fm  = lane & 15;
    const int fka = (lane >> 4) * 8;    // A: b128 loads at k = fka, fka+16
    const int fkb = (lane >> 4) * 8;    // B: b128 loads at k = fkb, fkb+16

    auto stage = [&](int k0, int buf) {
        const __bf16* ga = aRow + k0;
        async_copy_b128(ga,      oA[buf]);
        async_copy_b128(ga + 8,  oA[buf] + 16);
        async_copy_b128(ga + 16, oA[buf] + 32);
        async_copy_b128(ga + 24, oA[buf] + 48);
        const __bf16* gb = btRow + k0;
        async_copy_b128(gb,     oB[buf]);
        async_copy_b128(gb + 8, oB[buf] + 16);
    };

    stage(0, 0);
    wait_async0();
    __syncthreads();

    for (int it = 0; it < NITER; ++it) {
        const int buf = it & 1;
        if (it + 1 < NITER) stage((it + 1) * KT, buf ^ 1);   // fire-and-forget

        v16bf afr[4], bfr[4];
        #pragma unroll
        for (int mi = 0; mi < 4; ++mi) {
            const __bf16* p = &lA[buf][(wm + mi * 16 + fm) * LDSTR + fka];
            v8bf lo = *(const v8bf*)(p);
            v8bf hi = *(const v8bf*)(p + 16);
            afr[mi] = __builtin_shufflevector(lo, hi, 0,1,2,3,4,5,6,7,8,9,10,11,12,13,14,15);
        }
        #pragma unroll
        for (int ni = 0; ni < 4; ++ni) {
            const __bf16* p = &lBT[buf][(wn + ni * 16 + fm) * LDSTR + fkb];
            v8bf lo = *(const v8bf*)(p);
            v8bf hi = *(const v8bf*)(p + 16);
            bfr[ni] = __builtin_shufflevector(lo, hi, 0,1,2,3,4,5,6,7,8,9,10,11,12,13,14,15);
        }

        #pragma unroll
        for (int mi = 0; mi < 4; ++mi)
            #pragma unroll
            for (int ni = 0; ni < 4; ++ni)
                acc[mi][ni] = __builtin_amdgcn_wmma_f32_16x16x32_bf16(
                    false, afr[mi], false, bfr[ni],
                    (short)0, acc[mi][ni], false, false);

        if (it + 1 < NITER) wait_async0();   // async writes visible before barrier
        __syncthreads();
    }

    // ---- epilogue (C layout: col = lane&15, row = (lane>>4)*8 + r) ----
    #pragma unroll
    for (int mi = 0; mi < 4; ++mi) {
        #pragma unroll
        for (int ni = 0; ni < 4; ++ni) {
            const int col   = n0 + wn + ni * 16 + (lane & 15);
            const int rbase = m0 + wm + mi * 16 + (lane >> 4) * 8;
            if (MODE == 0) {
                const float bb = b1[col];
                const float cc = c2[col];
                #pragma unroll
                for (int r = 0; r < 8; ++r) {
                    float z = acc[mi][ni][r] + bb;
                    // 1 - tanh^2(z) = sech^2(z) = 4u/(1+u)^2, u = e^{-2|z|}  (branch-free)
                    float u = __expf(-2.0f * fabsf(z));
                    float op = 1.0f + u;
                    float sech2 = 4.0f * u / (op * op);
                    G[(size_t)(rbase + r) * DIM + col] = (__bf16)(sech2 * cc);
                }
            } else {
                #pragma unroll
                for (int r = 0; r < 8; ++r) {
                    int row  = rbase + r;
                    int bidx = row >> 12;            // / SEQ
                    int s    = row & (SEQ - 1);
                    const float* hb = h + (size_t)bidx * SEQ * DIM;
                    float ph = hb[(size_t)s * DIM + col];
                    int sp = (s + 1 < SEQ) ? s + 1 : SEQ - 1;
                    int sm = (s - 1 > 0)   ? s - 1 : 0;
                    float lap = hb[(size_t)sp * DIM + col]
                              + hb[(size_t)sm * DIM + col] - 2.0f * ph;
                    out[(size_t)row * DIM + col] = ph + 0.1f * (0.1f * lap - acc[mi][ni][r]);
                }
            }
        }
    }
}

// ---------------------------------------------------------------------------
extern "C" void kernel_launch(void* const* d_in, const int* in_sizes, int n_in,
                              void* d_out, int out_size, void* d_ws, size_t ws_size,
                              hipStream_t stream) {
    (void)in_sizes; (void)n_in; (void)out_size; (void)ws_size;

    const float* h  = (const float*)d_in[0];
    const float* W1 = (const float*)d_in[1];
    const float* b1 = (const float*)d_in[2];
    const float* W2 = (const float*)d_in[3];
    float* out = (float*)d_out;

    // workspace layout
    char* ws = (char*)d_ws;
    __bf16* W1bf = (__bf16*)(ws);                                     //  8 MiB
    __bf16* W1T  = (__bf16*)(ws + (size_t)DIM * DIM * 2);             //  8 MiB
    float*  c2   = (float*) (ws + (size_t)DIM * DIM * 4);             //  8 KiB
    __bf16* G    = (__bf16*)(ws + (size_t)DIM * DIM * 4 + 8192);      // 64 MiB
    __bf16* Hbf  = (__bf16*)(ws + (size_t)DIM * DIM * 4 + 8192
                                + (size_t)MTOT * DIM * 2);            // 64 MiB

    prep_h_kernel <<<(size_t)MTOT * DIM / (256 * 8), 256, 0, stream>>>(h, Hbf);
    prep_w1_kernel<<<(DIM * DIM) / 256, 256, 0, stream>>>(W1, W1bf, W1T);
    prep_c2_kernel<<<DIM / 256, 256, 0, stream>>>(W2, c2);

    dim3 grid(DIM / NT, MTOT / MT);   // 16 x 64
    // GEMM1: B = W1^T  -> BT(global) = W1 row-major  = W1bf
    gemm_kernel<0><<<grid, 256, 0, stream>>>(Hbf, W1bf, b1, c2, G, nullptr, nullptr);
    // GEMM2: B = W1    -> BT(global) = W1^T          = W1T
    gemm_kernel<1><<<grid, 256, 0, stream>>>(G, W1T, nullptr, nullptr, nullptr, h, out);
}
